// FCbasedGCN_48704929136872
// MI455X (gfx1250) — matrix operand order repaired and checked by reading, not compiled, not run on hardware
//
#include <hip/hip_runtime.h>
#include <hip/hip_bf16.h>

// ---------------------------------------------------------------------------
// FC-based GCN for MI455X (gfx1250, wave32, WMMA).
//
// Math reordering: agg@Wrel^T == scatter_add(x@Wrel^T), so we project to
// H=128 BEFORE the edge scatter, minimizing edge bytes (the roofline floor).
// GEMMs run on v_wmma_f32_16x16x32_f16 (f16 in, f32 accumulate).
// ---------------------------------------------------------------------------

typedef __attribute__((ext_vector_type(16))) _Float16 v16h;
typedef __attribute__((ext_vector_type(8)))  _Float16 v8h;
typedef __attribute__((ext_vector_type(8)))  float    v8f;

#define HDIM 128   // hidden channels (fixed by the reference)

// ---------------------------------------------------------------------------
// Convert fp32 -> f16 with zero padding in K (and optionally in rows).
// dst is [rows_dst, Kpad] f16; src is [rows_src, K] f32.
// ---------------------------------------------------------------------------
__global__ void cvt_pad_f16(const float* __restrict__ src, _Float16* __restrict__ dst,
                            int rows_src, long total, int K, int Kpad)
{
    long i = (long)blockIdx.x * blockDim.x + threadIdx.x;
    if (i >= total) return;
    int r = (int)(i / Kpad);
    int k = (int)(i % Kpad);
    float v = (r < rows_src && k < K) ? src[(size_t)r * K + k] : 0.0f;
    dst[i] = (_Float16)v;
}

// ---------------------------------------------------------------------------
// ReLU + convert fp32 [Npad,128] -> f16 written at column offset inside the
// [Npad,512] concat buffer (ldy = 512). Feeds both next layer and layer 5.
// ---------------------------------------------------------------------------
__global__ void relu_cvt_f16(const float* __restrict__ X, _Float16* __restrict__ Y,
                             int ldy, int total)
{
    int i = blockIdx.x * blockDim.x + threadIdx.x;
    if (i >= total) return;
    int r = i >> 7;         // / 128
    int c = i & 127;        // % 128
    float v = X[i];
    v = v > 0.0f ? v : 0.0f;
    Y[(size_t)r * ldy + c] = (_Float16)v;
}

// ---------------------------------------------------------------------------
// WMMA GEMM:  Out[m][n] (+bias[n]) = sum_k A_f16[m][k] * W_f16[n][k]
// A: [Npad, lda] f16 row-major (K contiguous)
// W: [128, ldw] f16 row-major  (K contiguous)  -> acts as B^T, perfect for WMMA
// Out: [Npad, 128] f32
// Block = 128 threads = 4 waves; each wave computes a 16-row x 128-col strip
// (8 accumulators of v8f). Grid.x = Npad/64.
//
// Lane layouts used by this code (contiguous k, same for A and B; not the ISA table order):
//   A 16x32 f16: lane L holds row M=L%16, K = (L<16?0:16)+{0..15}  (contig)
//   B 32x16 f16: lane L holds col N=L%16, K = (L<16?0:16)+{0..15}  (contig!)
//   C/D 16x16 f32: lane L, vgpr v -> (M = v + (L>=16?8:0), N = L%16)
// ---------------------------------------------------------------------------
__global__ __launch_bounds__(128)
void gemm_wmma_f16(const _Float16* __restrict__ A, int lda,
                   const _Float16* __restrict__ W, int ldw, int K,
                   const float* __restrict__ bias,
                   float* __restrict__ Out)
{
    const int  lane = threadIdx.x & 31;
    const int  wave = threadIdx.x >> 5;
    const int  row0 = blockIdx.x * 64 + wave * 16;
    const int  mlo  = lane & 15;
    const bool hi   = lane >= 16;

    const _Float16* arow = A + (size_t)(row0 + mlo) * lda + (hi ? 16 : 0);

    v8f acc[8] = {};

    for (int k0 = 0; k0 < K; k0 += 32) {
        // prefetch next A chunk (global_prefetch_b8; speculative, OOB is dropped)
        __builtin_prefetch(arow + k0 + 64, 0, 1);

        // A fragment: two contiguous 16B chunks per lane
        v8h alo = *(const v8h*)(arow + k0);
        v8h ahi = *(const v8h*)(arow + k0 + 8);
        v16h a;
#pragma unroll
        for (int i = 0; i < 8; ++i) { a[i] = alo[i]; a[i + 8] = ahi[i]; }

        const int kb = k0 + (hi ? 16 : 0);
#pragma unroll
        for (int ct = 0; ct < 8; ++ct) {
            // B fragment: one contiguous 32B chunk of weight row (= B column)
            const _Float16* wrow = W + (size_t)(ct * 16 + mlo) * ldw + kb;
            v16h b = *(const v16h*)wrow;
            acc[ct] = __builtin_amdgcn_wmma_f32_16x16x32_f16(
                false, a, false, b, (short)0, acc[ct], false, false);
        }
    }

    const int mbase = row0 + (hi ? 8 : 0);
#pragma unroll
    for (int ct = 0; ct < 8; ++ct) {
        const int n  = ct * 16 + mlo;
        const float bb = bias ? bias[n] : 0.0f;
#pragma unroll
        for (int v = 0; v < 8; ++v) {
            Out[(size_t)(mbase + v) * HDIM + n] = acc[ct][v] + bb;
        }
    }
}

// ---------------------------------------------------------------------------
// Edge aggregation: Out[dst] += P[src], H=128 floats per edge.
// 32 lanes per edge, float4 per lane -> fully coalesced 512B row moves.
// This is the bandwidth floor of the whole net (~1.6 GB over 5 layers).
// ---------------------------------------------------------------------------
__global__ void edge_agg(const float* __restrict__ P, const int* __restrict__ src,
                         const int* __restrict__ dst, float* __restrict__ Out, int E)
{
    int t = blockIdx.x * blockDim.x + threadIdx.x;
    int e = t >> 5;
    if (e >= E) return;
    int lane = t & 31;
    int s = src[e], d = dst[e];
    const float4 v = ((const float4*)(P + (size_t)s * HDIM))[lane];
    float* o = Out + (size_t)d * HDIM + lane * 4;
    atomicAdd(o + 0, v.x);
    atomicAdd(o + 1, v.y);
    atomicAdd(o + 2, v.z);
    atomicAdd(o + 3, v.w);
}

// ---------------------------------------------------------------------------
// BatchNorm statistics: per-channel sum and sum-of-squares over N real rows.
// ---------------------------------------------------------------------------
__global__ void bn_stats(const float* __restrict__ X, float* __restrict__ sums,
                         float* __restrict__ sqs, int N, int rows_per_block)
{
    int ch = threadIdx.x;                      // 128 channels
    int r0 = blockIdx.x * rows_per_block;
    int r1 = r0 + rows_per_block; if (r1 > N) r1 = N;
    float s = 0.0f, ss = 0.0f;
    for (int r = r0; r < r1; ++r) {
        float v = X[(size_t)r * HDIM + ch];
        s += v; ss += v * v;
    }
    atomicAdd(&sums[ch], s);
    atomicAdd(&sqs[ch], ss);
}

// ---------------------------------------------------------------------------
// Per-graph pooling sums (+counts) of raw x5 (BN folded into final linear).
// ---------------------------------------------------------------------------
__global__ void pool_sum(const float* __restrict__ X, const int* __restrict__ batch,
                         float* __restrict__ psum, float* __restrict__ pcnt, int N)
{
    int t = blockIdx.x * blockDim.x + threadIdx.x;
    int node = t >> 5;
    if (node >= N) return;
    int lane = t & 31;
    int g = batch[node];
    const float4 v = ((const float4*)(X + (size_t)node * HDIM))[lane];
    float* o = psum + (size_t)g * HDIM + lane * 4;
    atomicAdd(o + 0, v.x);
    atomicAdd(o + 1, v.y);
    atomicAdd(o + 2, v.z);
    atomicAdd(o + 3, v.w);
    if (lane == 0) atomicAdd(&pcnt[g], 1.0f);
}

// ---------------------------------------------------------------------------
// Final: out[g][c] = lb[c] + sum_h BN(psum[g][h]/cnt_g) * lw[c][h]
// BN commutes with mean-pool (per-channel affine), so fold it here.
// ---------------------------------------------------------------------------
__global__ void final_lin(const float* __restrict__ psum, const float* __restrict__ pcnt,
                          const float* __restrict__ sums, const float* __restrict__ sqs,
                          const float* __restrict__ gamma, const float* __restrict__ beta,
                          const float* __restrict__ lw, const float* __restrict__ lb,
                          float* __restrict__ out, int G, int C, int N)
{
    int t = blockIdx.x * blockDim.x + threadIdx.x;
    if (t >= G * C) return;
    int g = t / C, c = t % C;
    float cnt = pcnt[g]; cnt = cnt > 1.0f ? cnt : 1.0f;
    float invc = 1.0f / cnt;
    float invN = 1.0f / (float)N;
    float accv = lb[c];
    for (int h = 0; h < HDIM; ++h) {
        float mu  = sums[h] * invN;
        float var = sqs[h] * invN - mu * mu;
        float rs  = rsqrtf(var + 1e-5f);
        float xv  = (psum[(size_t)g * HDIM + h] * invc - mu) * rs * gamma[h] + beta[h];
        accv += xv * lw[(size_t)c * HDIM + h];
    }
    out[t] = accv;
}

// ---------------------------------------------------------------------------
extern "C" void kernel_launch(void* const* d_in, const int* in_sizes, int n_in,
                              void* d_out, int out_size, void* d_ws, size_t ws_size,
                              hipStream_t stream)
{
    (void)n_in; (void)ws_size;

    const float* x       = (const float*)d_in[0];
    const int*   eidx    = (const int*)d_in[1];
    const int*   batch   = (const int*)d_in[2];
    const float* w1_rel  = (const float*)d_in[3];
    const float* w1_root = (const float*)d_in[4];
    const float* b1      = (const float*)d_in[5];
    const float* w2_rel  = (const float*)d_in[6];
    const float* w2_root = (const float*)d_in[7];
    const float* b2      = (const float*)d_in[8];
    const float* w3_rel  = (const float*)d_in[9];
    const float* w3_root = (const float*)d_in[10];
    const float* b3      = (const float*)d_in[11];
    const float* w4_rel  = (const float*)d_in[12];
    const float* w4_root = (const float*)d_in[13];
    const float* b4      = (const float*)d_in[14];
    const float* w5_rel  = (const float*)d_in[15];
    const float* w5_root = (const float*)d_in[16];
    const float* b5      = (const float*)d_in[17];
    const float* gamma   = (const float*)d_in[18];
    const float* beta    = (const float*)d_in[19];
    const float* lin_w   = (const float*)d_in[20];
    const float* lin_b   = (const float*)d_in[21];

    const int H    = in_sizes[5];            // 128
    const int C    = in_sizes[21];           // 10
    const int FIN  = in_sizes[3] / H;        // 200
    const int N    = in_sizes[2];            // 20000
    const int E    = in_sizes[1] / 2;        // 320000
    const int G    = out_size / C;           // 100
    const int NPAD = ((N + 63) / 64) * 64;   // 20032 (64-row GEMM blocks)
    const int K1   = ((FIN + 31) / 32) * 32; // 224 (K padded to WMMA step)
    const int K4   = 4 * H;                  // 512

    const int* src = eidx;
    const int* dst = eidx + E;

    // ---- workspace carve (all 256B aligned) ----
    char* wsp = (char*)d_ws;
    auto carve = [&](size_t bytes) {
        char* p = wsp;
        wsp += (bytes + 255) & ~(size_t)255;
        return p;
    };
    _Float16* X0    = (_Float16*)carve((size_t)NPAD * K1 * 2);  // f16 input, K-padded
    _Float16* XC    = (_Float16*)carve((size_t)NPAD * K4 * 2);  // concat(x1..x4) f16
    float*    P     = (float*)   carve((size_t)NPAD * H * 4);   // rel-projection
    float*    OUT   = (float*)   carve((size_t)NPAD * H * 4);   // layer output / x5
    _Float16* W1R   = (_Float16*)carve((size_t)H * K1 * 2);
    _Float16* W1T   = (_Float16*)carve((size_t)H * K1 * 2);
    _Float16* W2R   = (_Float16*)carve((size_t)H * H * 2);
    _Float16* W2T   = (_Float16*)carve((size_t)H * H * 2);
    _Float16* W3R   = (_Float16*)carve((size_t)H * H * 2);
    _Float16* W3T   = (_Float16*)carve((size_t)H * H * 2);
    _Float16* W4R   = (_Float16*)carve((size_t)H * H * 2);
    _Float16* W4T   = (_Float16*)carve((size_t)H * H * 2);
    _Float16* W5R   = (_Float16*)carve((size_t)H * K4 * 2);
    _Float16* W5T   = (_Float16*)carve((size_t)H * K4 * 2);
    float*    sums  = (float*)   carve((size_t)H * 4);
    float*    sqs   = (float*)   carve((size_t)H * 4);
    float*    psum  = (float*)   carve((size_t)G * H * 4);
    float*    pcnt  = (float*)   carve((size_t)G * 4);

    auto cvt = [&](const float* s, _Float16* d, int rows_src, int rows_dst, int K, int Kpad) {
        long total = (long)rows_dst * Kpad;
        int blocks = (int)((total + 255) / 256);
        cvt_pad_f16<<<blocks, 256, 0, stream>>>(s, d, rows_src, total, K, Kpad);
    };

    // ---- weights + input to f16 (zero-padded in K) ----
    cvt(w1_rel,  W1R, H, H, FIN, K1);
    cvt(w1_root, W1T, H, H, FIN, K1);
    cvt(w2_rel,  W2R, H, H, H, H);
    cvt(w2_root, W2T, H, H, H, H);
    cvt(w3_rel,  W3R, H, H, H, H);
    cvt(w3_root, W3T, H, H, H, H);
    cvt(w4_rel,  W4R, H, H, H, H);
    cvt(w4_root, W4T, H, H, H, H);
    cvt(w5_rel,  W5R, H, H, K4, K4);
    cvt(w5_root, W5T, H, H, K4, K4);
    cvt(x, X0, N, NPAD, FIN, K1);

    const int ggrid   = NPAD / 64;
    const int agrid   = (E * 32 + 255) / 256;
    const int rgrid   = (NPAD * H + 255) / 256;
    const int rtotal  = NPAD * H;

    // ---- layer 1 (K=224): project, aggregate, relu+cvt into XC[:, 0:128] ----
    gemm_wmma_f16<<<ggrid, 128, 0, stream>>>(X0, K1, W1R, K1, K1, nullptr, P);
    gemm_wmma_f16<<<ggrid, 128, 0, stream>>>(X0, K1, W1T, K1, K1, b1, OUT);
    edge_agg<<<agrid, 256, 0, stream>>>(P, src, dst, OUT, E);
    relu_cvt_f16<<<rgrid, 256, 0, stream>>>(OUT, XC + 0 * H, K4, rtotal);

    // ---- layers 2..4 (K=128): input = previous column block of XC ----
    const _Float16* Wr[3] = {W2R, W3R, W4R};
    const _Float16* Wt[3] = {W2T, W3T, W4T};
    const float*    bb[3] = {b2, b3, b4};
    for (int l = 0; l < 3; ++l) {
        const _Float16* Ain = XC + (size_t)l * H;   // column block l, lda = 512
        gemm_wmma_f16<<<ggrid, 128, 0, stream>>>(Ain, K4, Wr[l], H, H, nullptr, P);
        gemm_wmma_f16<<<ggrid, 128, 0, stream>>>(Ain, K4, Wt[l], H, H, bb[l], OUT);
        edge_agg<<<agrid, 256, 0, stream>>>(P, src, dst, OUT, E);
        relu_cvt_f16<<<rgrid, 256, 0, stream>>>(OUT, XC + (size_t)(l + 1) * H, K4, rtotal);
    }

    // ---- layer 5 (K=512, input = full concat, no relu) -> OUT is x5 ----
    gemm_wmma_f16<<<ggrid, 128, 0, stream>>>(XC, K4, W5R, K4, K4, nullptr, P);
    gemm_wmma_f16<<<ggrid, 128, 0, stream>>>(XC, K4, W5T, K4, K4, b5, OUT);
    edge_agg<<<agrid, 256, 0, stream>>>(P, src, dst, OUT, E);

    // ---- BN stats + pooling + final linear (BN folded into linear) ----
    hipMemsetAsync(sums, 0, (size_t)H * 4, stream);
    hipMemsetAsync(sqs,  0, (size_t)H * 4, stream);
    hipMemsetAsync(psum, 0, (size_t)G * H * 4, stream);
    hipMemsetAsync(pcnt, 0, (size_t)G * 4, stream);

    const int rows_per_block = 160;
    bn_stats<<<(N + rows_per_block - 1) / rows_per_block, H, 0, stream>>>(
        OUT, sums, sqs, N, rows_per_block);
    pool_sum<<<(N * 32 + 255) / 256, 256, 0, stream>>>(OUT, batch, psum, pcnt, N);
    final_lin<<<(G * C + 255) / 256, 256, 0, stream>>>(
        psum, pcnt, sums, sqs, gamma, beta, lin_w, lin_b, (float*)d_out, G, C, N);
}
